// MultiHeadAttention_73907797229699
// MI455X (gfx1250) — compile-verified, matrix operand order checked by reading, not run on hardware
//
#include <hip/hip_runtime.h>
#include <hip/hip_bf16.h>

typedef __bf16 bf16x16 __attribute__((ext_vector_type(16)));
typedef float  f32x8   __attribute__((ext_vector_type(8)));

#define D_MODEL 1024
#define L_SEQ   2048
#define NHEADS  16
#define HDIM    64
#define BATCH   2
#define MTOT    (BATCH * L_SEQ)      // 4096 rows
#define BHTOT   (BATCH * NHEADS)     // 32 batch-heads
#define LOG2E   1.4426950408889634f

union FragU { struct { uint4 lo, hi; } q; bf16x16 v; };
union AccU  { f32x8 v; float f[8]; };

static __device__ __forceinline__ unsigned short f2b(float f) {
    unsigned u = __float_as_uint(f);
    u += 0x7FFFu + ((u >> 16) & 1u);   // round to nearest even
    return (unsigned short)(u >> 16);
}

static __device__ __forceinline__ f32x8 zero8() {
    AccU u;
#pragma unroll
    for (int i = 0; i < 8; ++i) u.f[i] = 0.f;
    return u.v;
}

static __device__ __forceinline__ f32x8 wmma_bf16(const bf16x16& a, const bf16x16& b, const f32x8& c) {
    return __builtin_amdgcn_wmma_f32_16x16x32_bf16(false, a, false, b, (short)0, c, false, false);
}

// Async global -> LDS copy, 16 bytes per lane (GLOBAL_LOAD_ASYNC_TO_LDS_B128).
// Generic shared-pointer low 32 bits == wave-relative LDS byte offset (ISA 10.2:
// LDS aperture lives in addr[63:32]).
static __device__ __forceinline__ void async_copy16(void* lds, const void* g) {
    unsigned ldsoff = (unsigned)(size_t)lds;
    asm volatile("global_load_async_to_lds_b128 %0, %1, off"
                 :: "v"(ldsoff), "v"(g) : "memory");
}

static __device__ __forceinline__ void async_wait0() {
#if __has_builtin(__builtin_amdgcn_s_wait_asynccnt)
    __builtin_amdgcn_s_wait_asynccnt(0);
#else
    asm volatile("s_wait_asynccnt 0x0" ::: "memory");
#endif
}

// ---------------------------------------------------------------------------
// fp32 -> bf16 elementwise cast (n multiple of 4)
// ---------------------------------------------------------------------------
__global__ void __launch_bounds__(256)
cast_f32_bf16(const float* __restrict__ in, unsigned short* __restrict__ out, int n) {
    int i = (blockIdx.x * blockDim.x + threadIdx.x) * 4;
    if (i < n) {
        float4 x = *(const float4*)(in + i);
        ushort4 y;
        y.x = f2b(x.x); y.y = f2b(x.y); y.z = f2b(x.z); y.w = f2b(x.w);
        *(ushort4*)(out + i) = y;
    }
}

// ---------------------------------------------------------------------------
// C = A * B^T   (A: [MTOT,1024] bf16 row-major, B: [1024,1024] bf16 row-major)
// MODE 0: store bf16 head-split [B, H, L, 64]   (for Q/K/V projections)
// MODE 1: store fp32 row-major  [MTOT, 1024]    (final output projection)
// Block: 256 thr (8 waves, 4x2), tile 128x128, K-step 64 (16 iters, 16 WMMA/iter).
// Double-buffered async global->LDS staging: next tile loads overlap compute.
// ---------------------------------------------------------------------------
template <int MODE>
__global__ void __launch_bounds__(256)
gemm_bt128(const unsigned short* __restrict__ A,
           const unsigned short* __restrict__ Bw,
           void* __restrict__ Cout) {
    __shared__ __align__(16) unsigned short As[2][128 * 72];  // pad 64->72: conflict-free b128
    __shared__ __align__(16) unsigned short Bs[2][128 * 72];

    const int tid  = threadIdx.x;
    const int lane = tid & 31;
    const int wid  = tid >> 5;
    const int wm   = wid & 3;          // 0..3 -> M offset *32
    const int wn   = wid >> 2;         // 0..1 -> N offset *64
    const int mbase = blockIdx.x * 128;
    const int nbase = blockIdx.y * 128;
    const int l15  = lane & 15;
    const int hi   = lane >> 4;

    f32x8 acc[2][4];
#pragma unroll
    for (int a = 0; a < 2; ++a)
#pragma unroll
        for (int b = 0; b < 4; ++b) acc[a][b] = zero8();

    const int srow = tid >> 1;         // 0..127
    const int scol = (tid & 1) * 32;   // 0 or 32
    const size_t rowA = (size_t)(mbase + srow) * D_MODEL + scol;
    const size_t rowB = (size_t)(nbase + srow) * D_MODEL + scol;

    // prologue: stage K-tile 0 into buffer 0
#pragma unroll
    for (int c = 0; c < 4; ++c) {
        async_copy16(&As[0][srow * 72 + scol + c * 8], A  + rowA + c * 8);
        async_copy16(&Bs[0][srow * 72 + scol + c * 8], Bw + rowB + c * 8);
    }

    int buf = 0;
    for (int k0 = 0; k0 < D_MODEL; k0 += 64) {
        async_wait0();          // tile(buf) fully in LDS (all waves after barrier)
        __syncthreads();

        if (k0 + 64 < D_MODEL) {   // issue next tile into buf^1, overlap with compute
            const unsigned short* gA = A  + rowA + k0 + 64;
            const unsigned short* gB = Bw + rowB + k0 + 64;
#pragma unroll
            for (int c = 0; c < 4; ++c) {
                async_copy16(&As[buf ^ 1][srow * 72 + scol + c * 8], gA + c * 8);
                async_copy16(&Bs[buf ^ 1][srow * 72 + scol + c * 8], gB + c * 8);
            }
        }

        const unsigned short* Ab = As[buf];
        const unsigned short* Bb = Bs[buf];
#pragma unroll
        for (int kk = 0; kk < 2; ++kk) {
            const int koff = kk * 32 + hi * 8;
            FragU af[2];
#pragma unroll
            for (int mt = 0; mt < 2; ++mt) {
                const unsigned short* p = &Ab[(wm * 32 + mt * 16 + l15) * 72 + koff];
                af[mt].q.lo = *(const uint4*)p;
                af[mt].q.hi = *(const uint4*)(p + 16);
            }
#pragma unroll
            for (int nt = 0; nt < 4; ++nt) {
                FragU bf;
                const unsigned short* p = &Bb[(wn * 64 + nt * 16 + l15) * 72 + koff];
                bf.q.lo = *(const uint4*)p;
                bf.q.hi = *(const uint4*)(p + 16);
#pragma unroll
                for (int mt = 0; mt < 2; ++mt)
                    acc[mt][nt] = wmma_bf16(af[mt].v, bf.v, acc[mt][nt]);
            }
        }
        buf ^= 1;
    }

#pragma unroll
    for (int mt = 0; mt < 2; ++mt)
#pragma unroll
        for (int nt = 0; nt < 4; ++nt) {
            AccU u; u.v = acc[mt][nt];
#pragma unroll
            for (int j = 0; j < 8; ++j) {
                int gm = mbase + wm * 32 + mt * 16 + j + hi * 8;
                int gn = nbase + wn * 64 + nt * 16 + l15;
                if (MODE == 0) {
                    int b = gm >> 11, l = gm & (L_SEQ - 1);
                    int h = gn >> 6,  d = gn & (HDIM - 1);
                    ((unsigned short*)Cout)[((size_t)(b * NHEADS + h) * L_SEQ + l) * HDIM + d] = f2b(u.f[j]);
                } else {
                    ((float*)Cout)[(size_t)gm * D_MODEL + gn] = u.f[j];
                }
            }
        }
}

// ---------------------------------------------------------------------------
// Flash attention: one block = (batch-head, 128 query rows), 8 waves x 16 rows.
// Streaming log2-domain softmax over 64 key tiles of 32 keys each.
// Double-buffered K/V staging (K async, V manual transpose); Q async in LDS.
// ---------------------------------------------------------------------------
__global__ void __launch_bounds__(256)
flash_attn(const unsigned short* __restrict__ Q,
           const unsigned short* __restrict__ K,
           const unsigned short* __restrict__ V,
           const int* __restrict__ amask,
           unsigned short* __restrict__ ctx) {
    __shared__ __align__(16) unsigned short Qs[128 * 72];       // q rows, pad 64->72
    __shared__ __align__(16) unsigned short Ks[2][32 * 72];     // key-major, pad 64->72
    __shared__ __align__(16) unsigned short Vt[2][64 * 40];     // d-major (transposed), pad 32->40
    __shared__ __align__(16) unsigned short Ps[8 * 16 * 40];    // per-wave P staging
    __shared__ float biasS[L_SEQ];                              // log2-domain mask bias

    const int tid  = threadIdx.x;
    const int lane = tid & 31;
    const int wid  = tid >> 5;
    const int l15  = lane & 15;
    const int hi   = lane >> 4;
    const int koff = hi * 8;
    const int bh   = blockIdx.y;
    const int b    = bh >> 4;
    const int h    = bh & 15;
    const int qbase = blockIdx.x * 128;
    const size_t base = (size_t)bh * L_SEQ * HDIM;

    const int srow = tid >> 3;        // 0..31 (key within tile)
    const int scol = (tid & 7) * 8;   // 0..56 (d)

    // prologue: Q tile (async), mask bias row, and K/V tile 0 into buffer 0
    {
        const int qrow0 = tid >> 1;          // 0..127
        const int qcol0 = (tid & 1) * 32;    // 0 or 32
        const unsigned short* gq = Q + base + (size_t)(qbase + qrow0) * HDIM + qcol0;
#pragma unroll
        for (int c = 0; c < 4; ++c)
            async_copy16(&Qs[qrow0 * 72 + qcol0 + c * 8], gq + c * 8);
#pragma unroll
        for (int i = tid; i < L_SEQ; i += 256)
            biasS[i] = (amask[b * L_SEQ + i] == 0) ? -1.0e9f * LOG2E : 0.f;

        const unsigned short* gk = K + base + (size_t)srow * HDIM + scol;
        async_copy16(&Ks[0][srow * 72 + scol], gk);
        const unsigned short* gv = V + base + (size_t)srow * HDIM + scol;
        uint4 vv = *(const uint4*)gv;
        const unsigned short* pv = (const unsigned short*)&vv;
#pragma unroll
        for (int e = 0; e < 8; ++e) Vt[0][(scol + e) * 40 + srow] = pv[e];
    }

    float m_run[8], l_run[8];
#pragma unroll
    for (int j = 0; j < 8; ++j) { m_run[j] = -3.0e38f; l_run[j] = 0.f; }
    f32x8 oacc[4];
#pragma unroll
    for (int dt = 0; dt < 4; ++dt) oacc[dt] = zero8();

    const unsigned short* qrow = &Qs[(wid * 16 + l15) * 72];
    const float sscale = 0.125f * LOG2E;   // 1/sqrt(64) folded with log2(e)

    int buf = 0;
    for (int kb = 0; kb < L_SEQ; kb += 32) {
        async_wait0();          // tile(buf) + Q ready after barrier
        __syncthreads();

        if (kb + 32 < L_SEQ) {  // stage next K/V tile into buf^1, overlap compute
            const unsigned short* gk = K + base + (size_t)(kb + 32 + srow) * HDIM + scol;
            async_copy16(&Ks[buf ^ 1][srow * 72 + scol], gk);
            const unsigned short* gv = V + base + (size_t)(kb + 32 + srow) * HDIM + scol;
            uint4 vv = *(const uint4*)gv;
            const unsigned short* pv = (const unsigned short*)&vv;
#pragma unroll
            for (int e = 0; e < 8; ++e) Vt[buf ^ 1][(scol + e) * 40 + srow] = pv[e];
        }

        const unsigned short* Kb = Ks[buf];
        const unsigned short* Vb = Vt[buf];

        // key-padding mask bias from LDS (log2 domain)
        float bias[2];
#pragma unroll
        for (int nt = 0; nt < 2; ++nt)
            bias[nt] = biasS[kb + nt * 16 + l15];

        // S = Q K^T  (16 x 32 scores): Q frags loaded transiently from LDS
        AccU s[2];
        {
            FragU qf, kf;
            qf.q.lo = *(const uint4*)(qrow + koff);            // d 0..31
            qf.q.hi = *(const uint4*)(qrow + koff + 16);
#pragma unroll
            for (int nt = 0; nt < 2; ++nt) {
                const unsigned short* p = &Kb[(nt * 16 + l15) * 72 + koff];
                kf.q.lo = *(const uint4*)p;
                kf.q.hi = *(const uint4*)(p + 16);
                s[nt].v = wmma_bf16(qf.v, kf.v, zero8());
            }
            qf.q.lo = *(const uint4*)(qrow + koff + 32);       // d 32..63
            qf.q.hi = *(const uint4*)(qrow + koff + 48);
#pragma unroll
            for (int nt = 0; nt < 2; ++nt) {
                const unsigned short* p = &Kb[(nt * 16 + l15) * 72 + koff];
                kf.q.lo = *(const uint4*)(p + 32);
                kf.q.hi = *(const uint4*)(p + 48);
                s[nt].v = wmma_bf16(qf.v, kf.v, s[nt].v);
            }
        }

        // streaming softmax (log2 domain): rows j (lanes 0-15) / j+8 (lanes 16-31)
        float alpha[8];
#pragma unroll
        for (int j = 0; j < 8; ++j) {
            float v0 = s[0].f[j] * sscale + bias[0];
            float v1 = s[1].f[j] * sscale + bias[1];
            float m = fmaxf(v0, v1);
            m = fmaxf(m, __shfl_xor(m, 1, 32));
            m = fmaxf(m, __shfl_xor(m, 2, 32));
            m = fmaxf(m, __shfl_xor(m, 4, 32));
            m = fmaxf(m, __shfl_xor(m, 8, 32));
            float nm = fmaxf(m_run[j], m);
            alpha[j] = exp2f(m_run[j] - nm);
            m_run[j] = nm;
            float p0 = exp2f(v0 - nm);
            float p1 = exp2f(v1 - nm);
            s[0].f[j] = p0; s[1].f[j] = p1;
            float rs = p0 + p1;
            rs += __shfl_xor(rs, 1, 32);
            rs += __shfl_xor(rs, 2, 32);
            rs += __shfl_xor(rs, 4, 32);
            rs += __shfl_xor(rs, 8, 32);
            l_run[j] = l_run[j] * alpha[j] + rs;
        }
#pragma unroll
        for (int dt = 0; dt < 4; ++dt) {
            AccU u; u.v = oacc[dt];
#pragma unroll
            for (int j = 0; j < 8; ++j) u.f[j] *= alpha[j];
            oacc[dt] = u.v;
        }

        // C-layout -> A-fragment layout for P via per-wave LDS staging
        unsigned short* pw = &Ps[wid * 16 * 40];
#pragma unroll
        for (int nt = 0; nt < 2; ++nt)
#pragma unroll
            for (int j = 0; j < 8; ++j)
                pw[(j + hi * 8) * 40 + nt * 16 + l15] = f2b(s[nt].f[j]);

        FragU pf;
        {
            const unsigned short* p = &pw[l15 * 40 + koff];
            pf.q.lo = *(const uint4*)p;
            pf.q.hi = *(const uint4*)(p + 16);
        }

        // O += P * V   (B-frags from transposed V tile: lane = d, contiguous keys)
#pragma unroll
        for (int dt = 0; dt < 4; ++dt) {
            FragU vf;
            const unsigned short* p = &Vb[(dt * 16 + l15) * 40 + koff];
            vf.q.lo = *(const uint4*)p;
            vf.q.hi = *(const uint4*)(p + 16);
            oacc[dt] = wmma_bf16(pf.v, vf.v, oacc[dt]);
        }
        buf ^= 1;
    }

    // normalize and write ctx in [B, L, H*64] bf16 layout (input of Wo GEMM)
#pragma unroll
    for (int dt = 0; dt < 4; ++dt) {
        AccU u; u.v = oacc[dt];
#pragma unroll
        for (int j = 0; j < 8; ++j) {
            int gq = qbase + wid * 16 + j + hi * 8;
            int d  = dt * 16 + l15;
            float o = u.f[j] / l_run[j];
            ctx[(size_t)(b * L_SEQ + gq) * D_MODEL + h * HDIM + d] = f2b(o);
        }
    }
}

// ---------------------------------------------------------------------------
extern "C" void kernel_launch(void* const* d_in, const int* in_sizes, int n_in,
                              void* d_out, int out_size, void* d_ws, size_t ws_size,
                              hipStream_t stream) {
    (void)in_sizes; (void)n_in; (void)out_size; (void)ws_size;

    const float* q  = (const float*)d_in[0];
    const float* k  = (const float*)d_in[1];
    const float* v  = (const float*)d_in[2];
    const int*   am = (const int*)  d_in[3];
    const float* Wq = (const float*)d_in[4];
    const float* Wk = (const float*)d_in[5];
    const float* Wv = (const float*)d_in[6];
    const float* Wo = (const float*)d_in[7];

    const size_t NA = (size_t)MTOT * D_MODEL;      // 4,194,304
    const size_t NW = (size_t)D_MODEL * D_MODEL;   // 1,048,576

    unsigned short* ws  = (unsigned short*)d_ws;   // total 64 MiB of ws used
    unsigned short* qb  = ws;
    unsigned short* kb  = qb  + NA;
    unsigned short* vb  = kb  + NA;
    unsigned short* Wqb = vb  + NA;
    unsigned short* Wkb = Wqb + NW;
    unsigned short* Wvb = Wkb + NW;
    unsigned short* Wob = Wvb + NW;
    unsigned short* Qh  = Wob + NW;
    unsigned short* Kh  = Qh  + NA;
    unsigned short* Vh  = Kh  + NA;
    unsigned short* ctx = Vh  + NA;

    // 1) fp32 -> bf16 casts
    cast_f32_bf16<<<dim3((unsigned)(NA / 1024)), dim3(256), 0, stream>>>(q,  qb,  (int)NA);
    cast_f32_bf16<<<dim3((unsigned)(NA / 1024)), dim3(256), 0, stream>>>(k,  kb,  (int)NA);
    cast_f32_bf16<<<dim3((unsigned)(NA / 1024)), dim3(256), 0, stream>>>(v,  vb,  (int)NA);
    cast_f32_bf16<<<dim3((unsigned)(NW / 1024)), dim3(256), 0, stream>>>(Wq, Wqb, (int)NW);
    cast_f32_bf16<<<dim3((unsigned)(NW / 1024)), dim3(256), 0, stream>>>(Wk, Wkb, (int)NW);
    cast_f32_bf16<<<dim3((unsigned)(NW / 1024)), dim3(256), 0, stream>>>(Wv, Wvb, (int)NW);
    cast_f32_bf16<<<dim3((unsigned)(NW / 1024)), dim3(256), 0, stream>>>(Wo, Wob, (int)NW);

    // 2) Q/K/V projections (x @ W^T) with fused head-split
    dim3 gg(MTOT / 128, D_MODEL / 128);
    gemm_bt128<0><<<gg, dim3(256), 0, stream>>>(qb, Wqb, Qh);
    gemm_bt128<0><<<gg, dim3(256), 0, stream>>>(kb, Wkb, Kh);
    gemm_bt128<0><<<gg, dim3(256), 0, stream>>>(vb, Wvb, Vh);

    // 3) flash attention
    flash_attn<<<dim3(L_SEQ / 128, BHTOT), dim3(256), 0, stream>>>(Qh, Kh, Vh, am, ctx);

    // 4) output projection (ctx @ Wo^T) -> fp32 d_out
    gemm_bt128<1><<<gg, dim3(256), 0, stream>>>(ctx, Wob, (float*)d_out);
}
